// MultiHeadAttention_2989297238586
// MI455X (gfx1250) — compile-verified, matrix operand order checked by reading, not run on hardware
//
#include <hip/hip_runtime.h>
#include <hip/hip_bf16.h>
#include <math.h>

typedef __attribute__((ext_vector_type(16))) __bf16         v16bf;
typedef __attribute__((ext_vector_type(8)))  float          v8f;
typedef __attribute__((ext_vector_type(8)))  unsigned short ushort8;

#define SEQ 4096
#define DM  768
#define NH  12
#define HD  64

// ---------- helpers ----------

static __device__ __forceinline__ unsigned short f2bf(float f) {
  union { float f; unsigned u; } v; v.f = f;
  unsigned u = v.u;
  u += 0x7FFFu + ((u >> 16) & 1u);   // round-to-nearest-even
  return (unsigned short)(u >> 16);
}

// Load a 16(M) x 32(K) bf16 tile in the CDNA5 WMMA A-operand layout.
// ISA 7.12.2 (16-bit A 16x32): lanes 0-15 hold M=lane, K={0..7,16..23};
// lanes 16-31 hold M=lane-16, K={8..15,24..31}. Each half is a contiguous
// 16-byte run in row-major memory -> two b128 loads per lane.
// The B operand (KxN) mirrors this with N<->lane, i.e. load B^T (NxK,
// K-major) with the same function.
static __device__ __forceinline__ v16bf load_tileA(const unsigned short* base, int stride) {
  const int lane = threadIdx.x & 31;
  const int m    = lane & 15;
  const int ko   = (lane >> 4) << 3;           // 0 or 8
  const unsigned short* p = base + (size_t)m * stride + ko;
  union { v16bf v; ushort8 h[2]; } u;
  u.h[0] = *(const ushort8*)(p);
  u.h[1] = *(const ushort8*)(p + 16);
  return u.v;
}

static __device__ __forceinline__ v16bf load_tileA_lds(const unsigned short* p) {
  const int lane = threadIdx.x & 31;
  const int m    = lane & 15;
  const int ko   = (lane >> 4) << 3;
  union { v16bf v; ushort8 h[2]; } u;
  u.h[0] = *(const ushort8*)(p + m * 32 + ko);
  u.h[1] = *(const ushort8*)(p + m * 32 + ko + 16);
  return u.v;
}

static __device__ __forceinline__ v8f wmma_bf16(v16bf a, v16bf b, v8f c) {
  // D = A(16x32) * B(32x16) + C, f32 accumulate
  return __builtin_amdgcn_wmma_f32_16x16x32_bf16(false, a, false, b, (short)0, c, false, false);
}

// ---------- kernels ----------

__global__ void __launch_bounds__(256) cvt_kernel(const float* __restrict__ src,
                                                  unsigned short* __restrict__ dst, int n) {
  int i = blockIdx.x * blockDim.x + threadIdx.x;
  if (i < n) dst[i] = f2bf(src[i]);
}

// One wave computes a 32(rows) x 64(head-dim) projection tile for one of
// {Q,K,V} and one head; applies RoPE (Q,K) or transposes (V) on the way out.
// Per K-step: 6 tile loads -> 8 WMMAs.
__global__ void __launch_bounds__(256) qkv_rope_kernel(
    const unsigned short* __restrict__ xbf,
    const unsigned short* __restrict__ wq,
    const unsigned short* __restrict__ wk,
    const unsigned short* __restrict__ wv,
    unsigned short* __restrict__ Q,     // (H, S, 64)
    unsigned short* __restrict__ Kh,    // (H, S, 64)
    unsigned short* __restrict__ Vt) {  // (H, 64, S)
  const int wid  = blockIdx.x * 8 + (threadIdx.x >> 5);
  const int lane = threadIdx.x & 31;
  const int rt   = wid & 127;          // 128 row tiles of 32
  const int mh   = wid >> 7;           // 0..35
  const int mat  = mh / NH;            // 0=Q 1=K 2=V
  const int h    = mh % NH;
  const int q0   = rt * 32;

  const unsigned short* W = (mat == 0) ? wq : (mat == 1) ? wk : wv;

  v8f acc[2][4] = {{v8f{}, v8f{}, v8f{}, v8f{}}, {v8f{}, v8f{}, v8f{}, v8f{}}};
  const unsigned short* Abase = xbf + (size_t)q0 * DM;
  const unsigned short* Bbase = W + (size_t)(h * HD) * DM;

  for (int k = 0; k < DM; k += 32) {
    if (k + 32 < DM) {
      __builtin_prefetch(Abase + k + 32, 0, 1);
      __builtin_prefetch(Abase + 16 * DM + k + 32, 0, 1);
    }
    v16bf a0 = load_tileA(Abase + k, DM);
    v16bf a1 = load_tileA(Abase + (size_t)16 * DM + k, DM);
#pragma unroll
    for (int t = 0; t < 4; ++t) {
      v16bf b = load_tileA(Bbase + (size_t)(t * 16) * DM + k, DM);
      acc[0][t] = wmma_bf16(a0, b, acc[0][t]);
      acc[1][t] = wmma_bf16(a1, b, acc[1][t]);
    }
  }

  const int nn = lane & 15;            // N within subtile
  const int mb = (lane >> 4) << 3;     // row-half offset (0 or 8)

  if (mat == 2) {
    // V: store transposed Vt[h*64 + n][s]; 8 consecutive s per lane -> b128
#pragma unroll
    for (int mt = 0; mt < 2; ++mt) {
#pragma unroll
      for (int t = 0; t < 4; ++t) {
        const int n = t * 16 + nn;
        ushort8 pk;
#pragma unroll
        for (int r = 0; r < 8; ++r) pk[r] = f2bf(acc[mt][t][r]);
        *(ushort8*)(Vt + (size_t)(h * HD + n) * SEQ + q0 + mt * 16 + mb) = pk;
      }
    }
  } else {
    // RoPE: pairs (2i, 2i+1) live in adjacent lanes; results go to i / 32+i
    unsigned short* dst = (mat == 0 ? Q : Kh) + ((size_t)h * SEQ + q0) * HD;
#pragma unroll
    for (int mt = 0; mt < 2; ++mt) {
#pragma unroll
      for (int t = 0; t < 4; ++t) {
        const int n   = t * 16 + nn;
        const int i   = n >> 1;
        const int odd = n & 1;
        const float inv = __powf(10000.0f, -(float)(2 * i) / 64.0f);
#pragma unroll
        for (int r = 0; r < 8; ++r) {
          const int   m  = mt * 16 + mb + r;
          const float sp = (float)(q0 + m);
          float c, sn;
          __sincosf(sp * inv, &sn, &c);
          const float mine  = acc[mt][t][r];
          const float other = __shfl_xor(mine, 1, 32);
          const float x1 = odd ? other : mine;
          const float x2 = odd ? mine  : other;
          const float res  = odd ? (x1 * sn + x2 * c) : (x1 * c - x2 * sn);
          const int   dout = odd ? (32 + i) : i;
          dst[(size_t)m * HD + dout] = f2bf(res);
        }
      }
    }
  }
}

// Flash attention, causal. One wave owns 32 queries x full head (hd=64),
// streams keys in chunks of 32 with online softmax, all matmuls on WMMA.
// Per 32-key chunk: 8 tile loads (K:4 + V:4) -> 16 WMMAs.
__global__ void __launch_bounds__(256) attn_kernel(
    const unsigned short* __restrict__ Q,    // (H, S, 64)
    const unsigned short* __restrict__ Kh,   // (H, S, 64)
    const unsigned short* __restrict__ Vt,   // (H, 64, S)
    unsigned short* __restrict__ O) {        // (S, 768) bf16
  __shared__ __align__(16) unsigned short pbuf[8][32 * 32];
  const int w    = threadIdx.x >> 5;
  const int lane = threadIdx.x & 31;
  const int h    = blockIdx.y;
  const int q0   = (blockIdx.x * 8 + w) * 32;
  const int nn   = lane & 15;
  const int mb   = (lane >> 4) << 3;

  const unsigned short* Qbase = Q + ((size_t)h * SEQ + q0) * HD;
  v16bf qa[2][2];
#pragma unroll
  for (int mt = 0; mt < 2; ++mt) {
    qa[mt][0] = load_tileA(Qbase + (size_t)(mt * 16) * HD, HD);       // hd 0..31
    qa[mt][1] = load_tileA(Qbase + (size_t)(mt * 16) * HD + 32, HD);  // hd 32..63
  }

  v8f o[2][4] = {{v8f{}, v8f{}, v8f{}, v8f{}}, {v8f{}, v8f{}, v8f{}, v8f{}}};
  float rmax[2][8], rsum[2][8];
#pragma unroll
  for (int mt = 0; mt < 2; ++mt)
#pragma unroll
    for (int r = 0; r < 8; ++r) { rmax[mt][r] = -3.0e38f; rsum[mt][r] = 0.0f; }

  unsigned short* pl = &pbuf[w][0];

  for (int k0 = 0; k0 < q0 + 32; k0 += 32) {
    // ---- S = Q K^T for 32 keys (K operand shared by both M-subtiles)
    const unsigned short* Kb = Kh + ((size_t)h * SEQ + k0) * HD;
    v16bf kb00 = load_tileA(Kb, HD);
    v16bf kb01 = load_tileA(Kb + 32, HD);
    v16bf kb10 = load_tileA(Kb + 16 * HD, HD);
    v16bf kb11 = load_tileA(Kb + 16 * HD + 32, HD);

#pragma unroll
    for (int mt = 0; mt < 2; ++mt) {
      v8f s0 = {}, s1 = {};
      s0 = wmma_bf16(qa[mt][0], kb00, s0);
      s0 = wmma_bf16(qa[mt][1], kb01, s0);
      s1 = wmma_bf16(qa[mt][0], kb10, s1);
      s1 = wmma_bf16(qa[mt][1], kb11, s1);

      // ---- online softmax; a row lives in one VGPR across a 16-lane half
#pragma unroll
      for (int r = 0; r < 8; ++r) {
        const int m = mt * 16 + mb + r;
        float a = s0[r] * 0.125f;                      // 1/sqrt(64)
        float b = s1[r] * 0.125f;
        if (k0 + nn      > q0 + m) a = -3.0e38f;       // causal mask
        if (k0 + 16 + nn > q0 + m) b = -3.0e38f;
        float mx = fmaxf(a, b);
        mx = fmaxf(mx, __shfl_xor(mx, 1, 32));
        mx = fmaxf(mx, __shfl_xor(mx, 2, 32));
        mx = fmaxf(mx, __shfl_xor(mx, 4, 32));
        mx = fmaxf(mx, __shfl_xor(mx, 8, 32));
        const float nm = fmaxf(rmax[mt][r], mx);
        const float sf = __expf(rmax[mt][r] - nm);
        rmax[mt][r] = nm;
        const float p0 = __expf(a - nm);
        const float p1 = __expf(b - nm);
        float ps = p0 + p1;
        ps += __shfl_xor(ps, 1, 32);
        ps += __shfl_xor(ps, 2, 32);
        ps += __shfl_xor(ps, 4, 32);
        ps += __shfl_xor(ps, 8, 32);
        rsum[mt][r] = rsum[mt][r] * sf + ps;
#pragma unroll
        for (int t = 0; t < 4; ++t) o[mt][t][r] *= sf;
        // stage P as bf16 row-major 32x32 for the A-layout reload
        pl[m * 32 + nn]      = f2bf(p0);
        pl[m * 32 + 16 + nn] = f2bf(p1);
      }
    }
    asm volatile("s_wait_dscnt 0" ::: "memory");
    v16bf pa0 = load_tileA_lds(pl);
    v16bf pa1 = load_tileA_lds(pl + 16 * 32);

    // ---- O += P V ; V^T is (hd, S) K-major, shared by both M-subtiles
    const unsigned short* Vb = Vt + (size_t)h * HD * SEQ + k0;
#pragma unroll
    for (int t = 0; t < 4; ++t) {
      v16bf vb = load_tileA(Vb + (size_t)(t * 16) * SEQ, SEQ);
      o[0][t] = wmma_bf16(pa0, vb, o[0][t]);
      o[1][t] = wmma_bf16(pa1, vb, o[1][t]);
    }
  }

  // ---- normalize and store O (S, 768) bf16
  unsigned short* Ob = O + (size_t)q0 * DM + h * HD;
#pragma unroll
  for (int mt = 0; mt < 2; ++mt) {
#pragma unroll
    for (int t = 0; t < 4; ++t) {
      const int d = t * 16 + nn;
#pragma unroll
      for (int r = 0; r < 8; ++r) {
        const int m = mt * 16 + mb + r;
        Ob[(size_t)m * DM + d] = f2bf(o[mt][t][r] / rsum[mt][r]);
      }
    }
  }
}

// out = O @ Wo^T + bo (fp32 output); 32x64 tile per wave
__global__ void __launch_bounds__(256) proj_kernel(
    const unsigned short* __restrict__ Ob,
    const unsigned short* __restrict__ wo,
    const float* __restrict__ bo,
    float* __restrict__ out) {
  const int wid  = blockIdx.x * 8 + (threadIdx.x >> 5);
  const int lane = threadIdx.x & 31;
  const int rt   = wid & 127;          // 128 row tiles of 32
  const int nt   = wid >> 7;           // 0..11
  const int q0   = rt * 32;
  const int n0   = nt * 64;

  v8f acc[2][4] = {{v8f{}, v8f{}, v8f{}, v8f{}}, {v8f{}, v8f{}, v8f{}, v8f{}}};
  const unsigned short* Abase = Ob + (size_t)q0 * DM;
  const unsigned short* Bbase = wo + (size_t)n0 * DM;
  for (int k = 0; k < DM; k += 32) {
    if (k + 32 < DM) {
      __builtin_prefetch(Abase + k + 32, 0, 1);
      __builtin_prefetch(Abase + 16 * DM + k + 32, 0, 1);
    }
    v16bf a0 = load_tileA(Abase + k, DM);
    v16bf a1 = load_tileA(Abase + (size_t)16 * DM + k, DM);
#pragma unroll
    for (int t = 0; t < 4; ++t) {
      v16bf b = load_tileA(Bbase + (size_t)(t * 16) * DM + k, DM);
      acc[0][t] = wmma_bf16(a0, b, acc[0][t]);
      acc[1][t] = wmma_bf16(a1, b, acc[1][t]);
    }
  }
  const int nn = lane & 15;
  const int mb = (lane >> 4) << 3;
#pragma unroll
  for (int mt = 0; mt < 2; ++mt) {
#pragma unroll
    for (int t = 0; t < 4; ++t) {
      const int n = n0 + t * 16 + nn;
      const float bias = bo[n];
#pragma unroll
      for (int r = 0; r < 8; ++r) {
        const int m = mt * 16 + mb + r;
        out[(size_t)(q0 + m) * DM + n] = acc[mt][t][r] + bias;   // lane-coalesced
      }
    }
  }
}

// ---------- launch ----------

extern "C" void kernel_launch(void* const* d_in, const int* in_sizes, int n_in,
                              void* d_out, int out_size, void* d_ws, size_t ws_size,
                              hipStream_t stream) {
  // inputs: x, mask(ignored; causal computed analytically), Wq, Wk, Wv, Wo, bo
  const float* x  = (const float*)d_in[0];
  const float* Wq = (const float*)d_in[2];
  const float* Wk = (const float*)d_in[3];
  const float* Wv = (const float*)d_in[4];
  const float* Wo = (const float*)d_in[5];
  const float* bo = (const float*)d_in[6];

  const size_t SZX = (size_t)SEQ * DM;   // 3,145,728
  const size_t SZW = (size_t)DM * DM;    //   589,824
  unsigned short* ws  = (unsigned short*)d_ws;   // ~36.2 MB total
  unsigned short* xbf = ws;
  unsigned short* wqb = xbf + SZX;
  unsigned short* wkb = wqb + SZW;
  unsigned short* wvb = wkb + SZW;
  unsigned short* wob = wvb + SZW;
  unsigned short* Qb  = wob + SZW;
  unsigned short* Kb  = Qb  + SZX;
  unsigned short* Vtb = Kb  + SZX;
  unsigned short* Ob  = Vtb + SZX;

  cvt_kernel<<<(int)((SZX + 255) / 256), 256, 0, stream>>>(x,  xbf, (int)SZX);
  cvt_kernel<<<(int)((SZW + 255) / 256), 256, 0, stream>>>(Wq, wqb, (int)SZW);
  cvt_kernel<<<(int)((SZW + 255) / 256), 256, 0, stream>>>(Wk, wkb, (int)SZW);
  cvt_kernel<<<(int)((SZW + 255) / 256), 256, 0, stream>>>(Wv, wvb, (int)SZW);
  cvt_kernel<<<(int)((SZW + 255) / 256), 256, 0, stream>>>(Wo, wob, (int)SZW);

  // 3 mats * 12 heads * 128 row-tiles = 4608 wave-tiles / 8 waves per block
  qkv_rope_kernel<<<576, 256, 0, stream>>>(xbf, wqb, wkb, wvb, Qb, Kb, Vtb);

  // 128 q-tiles of 32 per head / 8 waves per block
  attn_kernel<<<dim3(16, NH), 256, 0, stream>>>(Qb, Kb, Vtb, Ob);

  // 12 * 128 = 1536 wave-tiles / 8 = 192 blocks
  proj_kernel<<<192, 256, 0, stream>>>(Ob, wob, bo, (float*)d_out);
}
